// MultiheadedAttention_73005854097790
// MI455X (gfx1250) — matrix operand order repaired and checked by reading, not compiled, not run on hardware
//
#include <hip/hip_runtime.h>
#include <hip/hip_bf16.h>

// ---------------------------------------------------------------------------
// Multi-headed attention, N=8192, d_model=512, P=64, 8 identical heads.
// tile(head,8) @ w_o == head @ (sum of 8 row-blocks of w_o)   [64,512].
// All GEMMs via v_wmma_f32_16x16x32_f16 (f16 in, f32 accumulate).
// Flash-attention streaming with 4-way split-K + LSE merge for occupancy.
// K/V tiles staged to LDS with global_load_async_to_lds_b128 when available.
// ---------------------------------------------------------------------------

typedef _Float16 h16;
typedef __attribute__((ext_vector_type(16))) _Float16 v16h;
typedef __attribute__((ext_vector_type(2)))  _Float16 v2h;
typedef __attribute__((ext_vector_type(8)))  float    v8f;
typedef __attribute__((ext_vector_type(4)))  int      v4i;

#define NWORDS 8192
#define DMODEL 512
#define PDIM   64
#define SEG    4                 // split-K segments over the key dimension
#define KBLK   (NWORDS / 64)     // 128 key blocks of 64
#define KBSEG  (KBLK / SEG)      // 32 key blocks per segment

#define AS1 __attribute__((address_space(1)))
#define AS3 __attribute__((address_space(3)))

__device__ __forceinline__ v8f wmma16(v16h a, v16h b, v8f c) {
  return __builtin_amdgcn_wmma_f32_16x16x32_f16(
      /*neg_a=*/false, a, /*neg_b=*/false, b,
      /*c_mod=*/(short)0, c, /*reuse_a=*/false, /*reuse_b=*/false);
}

// 16-byte global -> LDS async copy (ASYNCcnt path) with portable fallback.
__device__ __forceinline__ void async_b128(const void* g, void* l) {
#if __has_builtin(__builtin_amdgcn_global_load_async_to_lds_b128)
  void* gnc = const_cast<void*>(g);
  __builtin_amdgcn_global_load_async_to_lds_b128(
      (AS1 v4i*)gnc, (AS3 v4i*)l, 0, 0);
#else
  *(uint4*)l = *(const uint4*)g;
#endif
}

__device__ __forceinline__ void wait_async() {
#if __has_builtin(__builtin_amdgcn_global_load_async_to_lds_b128)
#if __has_builtin(__builtin_amdgcn_s_wait_asynccnt)
  __builtin_amdgcn_s_wait_asynccnt(0);
#else
  asm volatile("s_wait_asynccnt 0" ::: "memory");
#endif
#endif
}

// A-fragment (16x32 f16, row-major source, leading dim ld).
// ISA layout: lane = hf*16 + m ; (vgpr j, slot t): K = 16g + 8*hf + 2jj + t
__device__ __forceinline__ v16h load_a_f16(const h16* base, int ld, int k0) {
  const int lane = threadIdx.x & 31;
  const int m = lane & 15, hf = lane >> 4;
  const h16* row = base + (size_t)m * ld + k0 + 8 * hf;
  v16h a;
#pragma unroll
  for (int j = 0; j < 8; ++j) {
    const int g = j >> 2, jj = j & 3;
    v2h p = *(const v2h*)(row + 16 * g + 2 * jj);
    a[2 * j] = p[0];
    a[2 * j + 1] = p[1];
  }
  return a;
}

// B-fragment (32x16) from n-major f16 storage bt[n][k] (leading dim ld).
// ISA layout: lane = hf*16 + n ; (vgpr j, slot t): K = 16g + 8*hf + 2jj + t
__device__ __forceinline__ v16h load_b_nmajor(const h16* bt, int ld, int n0, int k0) {
  const int lane = threadIdx.x & 31;
  const int n = lane & 15, hf = lane >> 4;
  const h16* col = bt + (size_t)(n0 + n) * ld + k0 + 8 * hf;
  v16h b;
#pragma unroll
  for (int j = 0; j < 8; ++j) {
    v2h p = *(const v2h*)(col + 16 * (j >> 2) + 2 * (j & 3));
    b[2 * j] = p[0];
    b[2 * j + 1] = p[1];
  }
  return b;
}

// ---------------------------------------------------------------------------
// Prep A: convert x (fp32) to f16 once.
// ---------------------------------------------------------------------------
__global__ void __launch_bounds__(256)
xcvt_kernel(const float* __restrict__ x, h16* __restrict__ x16) {
  const int idx = blockIdx.x * blockDim.x + threadIdx.x;  // pairs
  if (idx >= NWORDS * DMODEL / 2) return;
  const float2 f = ((const float2*)x)[idx];
  v2h p; p[0] = (h16)f.x; p[1] = (h16)f.y;
  *(v2h*)(x16 + 2 * (size_t)idx) = p;
}

// ---------------------------------------------------------------------------
// Prep B: f16 n-major transposes of w_q/w_k/w_v ([64][512]) and
//         W_sum^T[c][r] = sum_h w_o[h*64+r][c]  ([512][64]).
// ---------------------------------------------------------------------------
__global__ void __launch_bounds__(256)
wprep_kernel(const float* __restrict__ wq, const float* __restrict__ wk,
             const float* __restrict__ wv, const float* __restrict__ wo,
             h16* __restrict__ wqt, h16* __restrict__ wkt,
             h16* __restrict__ wvt, h16* __restrict__ wsumt) {
  const int idx = blockIdx.x * blockDim.x + threadIdx.x;
  if (idx >= PDIM * DMODEL) return;
  // wt[n][k] = w[k][n]
  const int n = idx >> 9;            // 0..63
  const int k = idx & (DMODEL - 1);  // 0..511
  wqt[(size_t)n * DMODEL + k] = (h16)wq[(size_t)k * PDIM + n];
  wkt[(size_t)n * DMODEL + k] = (h16)wk[(size_t)k * PDIM + n];
  wvt[(size_t)n * DMODEL + k] = (h16)wv[(size_t)k * PDIM + n];
  // wsumt[c][r]
  const int c = idx >> 6;   // 0..511
  const int r = idx & 63;   // 0..63
  float s = 0.f;
#pragma unroll
  for (int hh = 0; hh < 8; ++hh) s += wo[(size_t)(hh * 64 + r) * DMODEL + c];
  wsumt[(size_t)c * PDIM + r] = (h16)s;
}

// ---------------------------------------------------------------------------
// Phase 1: Q,K,V = x @ w_{q,k,v} -> f16 [8192,64]; Q pre-scaled by 1/sqrt(64)
// grid 512, block 128 (4 waves; wave w owns output columns [16w,16w+16))
// ---------------------------------------------------------------------------
__global__ void __launch_bounds__(128)
qkv_kernel(const h16* __restrict__ x16, const h16* __restrict__ wqt,
           const h16* __restrict__ wkt, const h16* __restrict__ wvt,
           h16* __restrict__ q16, h16* __restrict__ k16, h16* __restrict__ v16) {
  const int rb = blockIdx.x;
  const int wave = threadIdx.x >> 5;
  const int n0 = wave * 16;
  const h16* xt = x16 + (size_t)rb * 16 * DMODEL;

  v8f cq = {}, ck = {}, cv = {};
  for (int ks = 0; ks < DMODEL / 32; ++ks) {
    const int k0 = ks * 32;
    v16h a = load_a_f16(xt, DMODEL, k0);
    cq = wmma16(a, load_b_nmajor(wqt, DMODEL, n0, k0), cq);
    ck = wmma16(a, load_b_nmajor(wkt, DMODEL, n0, k0), ck);
    cv = wmma16(a, load_b_nmajor(wvt, DMODEL, n0, k0), cv);
  }
  const int lane = threadIdx.x & 31;
  const int n = lane & 15, hf = lane >> 4;
#pragma unroll
  for (int r = 0; r < 8; ++r) {
    const size_t row = (size_t)rb * 16 + r + 8 * hf;
    q16[row * PDIM + n0 + n] = (h16)(cq[r] * 0.125f);  // fold 1/sqrt(64)
    k16[row * PDIM + n0 + n] = (h16)ck[r];
    v16[row * PDIM + n0 + n] = (h16)cv[r];
  }
}

// ---------------------------------------------------------------------------
// Phase 2: split-K flash attention.
// grid (128, SEG): x = 64-query-row block, y = key segment (2048 keys each).
// block 128 (4 waves, 16 rows/wave). Writes unnormalized O + per-row (m, l).
// ---------------------------------------------------------------------------
__global__ void __launch_bounds__(128)
flash_kernel(const h16* __restrict__ q16, const h16* __restrict__ k16,
             const h16* __restrict__ v16, float* __restrict__ po,
             float* __restrict__ pm, float* __restrict__ pl) {
  __shared__ __align__(16) h16 kt[64 * 64];       // [key][dim]
  __shared__ __align__(16) h16 vrow[64 * 64];     // [key][dim] staging
  __shared__ __align__(16) h16 vt[64 * 64];       // [dim][key]
  __shared__ __align__(16) h16 ps[4 * 16 * 64];   // per-wave P scratch

  const int wave = threadIdx.x >> 5;
  const int lane = threadIdx.x & 31;
  const int n = lane & 15, hf = lane >> 4;
  const int seg = blockIdx.y;
  const int qrow0 = blockIdx.x * 64 + wave * 16;

  const v16h aq0 = load_a_f16(q16 + (size_t)qrow0 * PDIM, PDIM, 0);
  const v16h aq1 = load_a_f16(q16 + (size_t)qrow0 * PDIM, PDIM, 32);

  v8f o0 = {}, o1 = {}, o2 = {}, o3 = {};
  float mrow[8], lrow[8];
#pragma unroll
  for (int r = 0; r < 8; ++r) { mrow[r] = -1e30f; lrow[r] = 0.f; }

  h16* myps = ps + wave * 16 * 64;

  const int kb_beg = seg * KBSEG, kb_end = kb_beg + KBSEG;
  for (int kb = kb_beg; kb < kb_end; ++kb) {
    __syncthreads();
    {  // async-stage K tile (final layout) and V tile (row-major staging)
      const h16* ksrc = k16 + (size_t)kb * 64 * PDIM;
      const h16* vsrc = v16 + (size_t)kb * 64 * PDIM;
      for (int i = threadIdx.x; i < (64 * PDIM) / 8; i += 128) {
        async_b128(ksrc + (size_t)i * 8, kt + (size_t)i * 8);
        async_b128(vsrc + (size_t)i * 8, vrow + (size_t)i * 8);
      }
      if (kb + 1 < kb_end) {
        __builtin_prefetch(k16 + (size_t)(kb + 1) * 64 * PDIM, 0, 0);
        __builtin_prefetch(v16 + (size_t)(kb + 1) * 64 * PDIM, 0, 0);
      }
      wait_async();
    }
    __syncthreads();
    // LDS->LDS transpose of V into [dim][key]
    for (int i = threadIdx.x; i < 64 * PDIM; i += 128)
      vt[(i & 63) * 64 + (i >> 6)] = vrow[i];
    __syncthreads();

    // S = Q K^T for this wave's 16 rows x 64 keys (4 n-subtiles, K=64)
    v8f s0 = {}, s1 = {}, s2 = {}, s3 = {};
    s0 = wmma16(aq0, load_b_nmajor(kt, PDIM, 0, 0), s0);
    s0 = wmma16(aq1, load_b_nmajor(kt, PDIM, 0, 32), s0);
    s1 = wmma16(aq0, load_b_nmajor(kt, PDIM, 16, 0), s1);
    s1 = wmma16(aq1, load_b_nmajor(kt, PDIM, 16, 32), s1);
    s2 = wmma16(aq0, load_b_nmajor(kt, PDIM, 32, 0), s2);
    s2 = wmma16(aq1, load_b_nmajor(kt, PDIM, 32, 32), s2);
    s3 = wmma16(aq0, load_b_nmajor(kt, PDIM, 48, 0), s3);
    s3 = wmma16(aq1, load_b_nmajor(kt, PDIM, 48, 32), s3);

    // Online softmax. Lane holds rows r+8*hf; a row spans the 16 lanes of a
    // half-group, so xor-shuffles 1,2,4,8 implement the row reduction.
    float sc[8];
#pragma unroll
    for (int r = 0; r < 8; ++r) {
      float m = fmaxf(fmaxf(s0[r], s1[r]), fmaxf(s2[r], s3[r]));
#pragma unroll
      for (int mask = 1; mask < 16; mask <<= 1) m = fmaxf(m, __shfl_xor(m, mask, 32));
      const float mnew = fmaxf(mrow[r], m);
      const float scale = __expf(mrow[r] - mnew);
      const float p0 = __expf(s0[r] - mnew);
      const float p1 = __expf(s1[r] - mnew);
      const float p2 = __expf(s2[r] - mnew);
      const float p3 = __expf(s3[r] - mnew);
      float ls = (p0 + p1) + (p2 + p3);
#pragma unroll
      for (int mask = 1; mask < 16; mask <<= 1) ls += __shfl_xor(ls, mask, 32);
      lrow[r] = lrow[r] * scale + ls;
      mrow[r] = mnew;
      sc[r] = scale;
      const int row = r + 8 * hf;
      myps[row * 64 + 0 * 16 + n] = (h16)p0;   // D-layout -> LDS (A reload)
      myps[row * 64 + 1 * 16 + n] = (h16)p1;
      myps[row * 64 + 2 * 16 + n] = (h16)p2;
      myps[row * 64 + 3 * 16 + n] = (h16)p3;
    }
#pragma unroll
    for (int r = 0; r < 8; ++r) {
      o0[r] *= sc[r]; o1[r] *= sc[r]; o2[r] *= sc[r]; o3[r] *= sc[r];
    }

    // O += P V  (P as A-fragments from per-wave LDS; B from transposed V)
    const v16h ap0 = load_a_f16(myps, 64, 0);
    const v16h ap1 = load_a_f16(myps, 64, 32);
    o0 = wmma16(ap0, load_b_nmajor(vt, 64, 0, 0), o0);
    o0 = wmma16(ap1, load_b_nmajor(vt, 64, 0, 32), o0);
    o1 = wmma16(ap0, load_b_nmajor(vt, 64, 16, 0), o1);
    o1 = wmma16(ap1, load_b_nmajor(vt, 64, 16, 32), o1);
    o2 = wmma16(ap0, load_b_nmajor(vt, 64, 32, 0), o2);
    o2 = wmma16(ap1, load_b_nmajor(vt, 64, 32, 32), o2);
    o3 = wmma16(ap0, load_b_nmajor(vt, 64, 48, 0), o3);
    o3 = wmma16(ap1, load_b_nmajor(vt, 64, 48, 32), o3);
  }

  // Write unnormalized partials + row stats for the LSE merge.
  float* poseg = po + (size_t)seg * NWORDS * PDIM;
#pragma unroll
  for (int r = 0; r < 8; ++r) {
    const size_t row = (size_t)qrow0 + r + 8 * hf;
    poseg[row * PDIM + 0 * 16 + n] = o0[r];
    poseg[row * PDIM + 1 * 16 + n] = o1[r];
    poseg[row * PDIM + 2 * 16 + n] = o2[r];
    poseg[row * PDIM + 3 * 16 + n] = o3[r];
    if (n == 0) {
      pm[(size_t)seg * NWORDS + row] = mrow[r];
      pl[(size_t)seg * NWORDS + row] = lrow[r];
    }
  }
}

// ---------------------------------------------------------------------------
// Phase 3: merge split-K partials with log-sum-exp weights -> head16.
// ---------------------------------------------------------------------------
__global__ void __launch_bounds__(256)
merge_kernel(const float* __restrict__ po, const float* __restrict__ pm,
             const float* __restrict__ pl, h16* __restrict__ head16) {
  const int idx = blockIdx.x * blockDim.x + threadIdx.x;
  if (idx >= NWORDS * PDIM) return;
  const int row = idx >> 6;
  float M = -1e30f;
#pragma unroll
  for (int s = 0; s < SEG; ++s) M = fmaxf(M, pm[(size_t)s * NWORDS + row]);
  float L = 0.f, acc = 0.f;
#pragma unroll
  for (int s = 0; s < SEG; ++s) {
    const float w = __expf(pm[(size_t)s * NWORDS + row] - M);
    L += pl[(size_t)s * NWORDS + row] * w;
    acc += po[(size_t)s * NWORDS * PDIM + idx] * w;
  }
  head16[idx] = (h16)(acc / L);
}

// ---------------------------------------------------------------------------
// Phase 4: out[8192,512] = head[8192,64] @ W_sum[64,512]   (fp32 output)
// grid 512 (16 rows/WG), block 128 (wave w owns cols [128w,128w+128))
// ---------------------------------------------------------------------------
__global__ void __launch_bounds__(128)
out_kernel(const h16* __restrict__ head16, const h16* __restrict__ wsumt,
           float* __restrict__ out) {
  const int rb = blockIdx.x;
  const int wave = threadIdx.x >> 5;
  const int lane = threadIdx.x & 31;
  const int n = lane & 15, hf = lane >> 4;
  const int row0 = rb * 16;

  const v16h a0 = load_a_f16(head16 + (size_t)row0 * PDIM, PDIM, 0);
  const v16h a1 = load_a_f16(head16 + (size_t)row0 * PDIM, PDIM, 32);

#pragma unroll
  for (int st = 0; st < 8; ++st) {
    const int n0 = wave * 128 + st * 16;
    v8f c = {};
    c = wmma16(a0, load_b_nmajor(wsumt, PDIM, n0, 0), c);
    c = wmma16(a1, load_b_nmajor(wsumt, PDIM, n0, 32), c);
#pragma unroll
    for (int r = 0; r < 8; ++r)
      out[((size_t)row0 + r + 8 * hf) * DMODEL + n0 + n] = c[r];
  }
}

// ---------------------------------------------------------------------------
extern "C" void kernel_launch(void* const* d_in, const int* in_sizes, int n_in,
                              void* d_out, int out_size, void* d_ws, size_t ws_size,
                              hipStream_t stream) {
  (void)in_sizes; (void)n_in; (void)out_size; (void)ws_size;
  const float* x  = (const float*)d_in[0];
  const float* wq = (const float*)d_in[1];
  const float* wk = (const float*)d_in[2];
  const float* wv = (const float*)d_in[3];
  const float* wo = (const float*)d_in[4];
  float* out = (float*)d_out;

  // Workspace layout (~20.5 MB total).
  h16* x16    = (h16*)d_ws;                          // 8192*512
  h16* q16    = x16 + (size_t)NWORDS * DMODEL;       // 8192*64
  h16* k16    = q16 + (size_t)NWORDS * PDIM;
  h16* v16    = k16 + (size_t)NWORDS * PDIM;
  h16* head16 = v16 + (size_t)NWORDS * PDIM;
  h16* wqt    = head16 + (size_t)NWORDS * PDIM;      // 64*512 each
  h16* wkt    = wqt + (size_t)PDIM * DMODEL;
  h16* wvt    = wkt + (size_t)PDIM * DMODEL;
  h16* wsumt  = wvt + (size_t)PDIM * DMODEL;         // 512*64
  float* po   = (float*)(wsumt + (size_t)PDIM * DMODEL);  // SEG*8192*64
  float* pm   = po + (size_t)SEG * NWORDS * PDIM;         // SEG*8192
  float* pl   = pm + (size_t)SEG * NWORDS;                // SEG*8192

  xcvt_kernel<<<(NWORDS * DMODEL / 2 + 255) / 256, 256, 0, stream>>>(x, x16);
  wprep_kernel<<<(PDIM * DMODEL + 255) / 256, 256, 0, stream>>>(
      wq, wk, wv, wo, wqt, wkt, wvt, wsumt);
  qkv_kernel<<<NWORDS / 16, 128, 0, stream>>>(x16, wqt, wkt, wvt, q16, k16, v16);
  flash_kernel<<<dim3(NWORDS / 64, SEG), 128, 0, stream>>>(q16, k16, v16, po, pm, pl);
  merge_kernel<<<(NWORDS * PDIM + 255) / 256, 256, 0, stream>>>(po, pm, pl, head16);
  out_kernel<<<NWORDS / 16, 128, 0, stream>>>(head16, wsumt, out);
}
